// AdaGAE_39127152066566
// MI455X (gfx1250) — matrix operand order repaired and checked by reading, not compiled, not run on hardware
//
#include <hip/hip_runtime.h>
#include <math.h>

// ---------------- problem constants (match reference) ----------------
#define NN      8192
#define D_IN    1024
#define D_MID   256
#define D_EMB   64
#define N_EDGES (NN * 8)

typedef __attribute__((ext_vector_type(16))) _Float16 v16h;
typedef __attribute__((ext_vector_type(8)))  float    v8f;

// ---------------------------------------------------------------------
// f16 fragment builders for V_WMMA_F32_16X16X32_F16 (CDNA5 ISA 7.12.2).
// All sources are pre-converted f16, so each lane's fragment is one or
// two contiguous 16B runs -> global_load_b128, no VALU cvt in hot loops.
//
// A (16x32, MxK): lanes 0-15: M=lane, K=0..15;
//                 lanes 16-31: M=lane-16, K=16..31.
// ---------------------------------------------------------------------
__device__ __forceinline__ v16h load_a_frag_h(const _Float16* __restrict__ src,
                                              int ld, int row0, int kb, int lane) {
    const int m    = lane & 15;
    const int koff = (lane & 16) ? 16 : 0;
    const _Float16* p = src + (size_t)(row0 + m) * ld + kb + koff;
    v16h a;
#pragma unroll
    for (int i = 0; i < 8; ++i) {        // frag idx i -> K=koff+i
        a[i]     = p[i];
        a[8 + i] = p[8 + i];
    }
    return a;
}

// B (32x16, KxN): lanes 0-15: N=lane, K=0..15; lanes 16-31: N=lane-16, K=16..31.
// Source is stored K-transposed: srcT[N][K] row-major, so B[k][n]=srcT[col0+n][kb+k]
// and each lane reads 16 contiguous halves (32B).
__device__ __forceinline__ v16h load_b_frag_h(const _Float16* __restrict__ srcT,
                                              int ld, int kb, int col0, int lane) {
    const int n    = lane & 15;
    const int koff = (lane & 16) ? 16 : 0;
    const _Float16* p = srcT + (size_t)(col0 + n) * ld + kb + koff;
    v16h b;
#pragma unroll
    for (int i = 0; i < 16; ++i) b[i] = p[i];
    return b;
}

// ---------------------------------------------------------------------
// precision-conversion helpers (run once per operand, off the hot path)
// ---------------------------------------------------------------------
__global__ void cvt_f32_to_f16(const float* __restrict__ src,
                               _Float16* __restrict__ dst, int n, int do_relu) {
    int i = blockIdx.x * blockDim.x + threadIdx.x;
    if (i < n) {
        float x = src[i];
        if (do_relu) x = fmaxf(x, 0.0f);
        dst[i] = (_Float16)x;
    }
}

// dst[n][K] (f16) = src[K][n] (f32): grid = K blocks, block = Nc threads
__global__ void cvt_transpose_f16(const float* __restrict__ src,
                                  _Float16* __restrict__ dst, int K, int Nc) {
    int k = blockIdx.x;
    int n = threadIdx.x;
    dst[(size_t)n * K + k] = (_Float16)src[(size_t)k * Nc + n];
}

// ---------------------------------------------------------------------
// GEMM1: Y[8192,256] = Xh[8192,1024] @ W1 (B from W1Th[256][1024])
// one wave per 16x16 tile, K swept 32/step
// ---------------------------------------------------------------------
__global__ void gemm_x_w1(const _Float16* __restrict__ Xh,
                          const _Float16* __restrict__ W1Th,
                          float* __restrict__ Y) {
    const int lane = threadIdx.x & 31;
    const int gw   = blockIdx.x * 8 + (threadIdx.x >> 5);  // 8192 tiles
    const int tn   = gw & 15;
    const int tm   = gw >> 4;
    const int row0 = tm * 16, col0 = tn * 16;

    v8f c = {};
#pragma unroll 4
    for (int kb = 0; kb < D_IN; kb += 32) {
        v16h a = load_a_frag_h(Xh, D_IN, row0, kb, lane);
        v16h b = load_b_frag_h(W1Th, D_IN, kb, col0, lane);
        c = __builtin_amdgcn_wmma_f32_16x16x32_f16(false, a, false, b,
                                                   (short)0, c, false, false);
    }
    const int mbase = (lane & 16) ? 8 : 0;
    const int n     = lane & 15;
#pragma unroll
    for (int r = 0; r < 8; ++r)
        Y[(size_t)(row0 + mbase + r) * D_MID + col0 + n] = c[r];
}

// ---------------------------------------------------------------------
// GEMM2: Z[8192,64] = Hh[8192,256] @ W2 (B from W2Th[64][256]); ReLU was
// fused into the H1->Hh conversion.
// ---------------------------------------------------------------------
__global__ void gemm_h_w2(const _Float16* __restrict__ Hh,
                          const _Float16* __restrict__ W2Th,
                          float* __restrict__ Z) {
    const int lane = threadIdx.x & 31;
    const int gw   = blockIdx.x * 8 + (threadIdx.x >> 5);  // 2048 tiles
    const int tn   = gw & 3;
    const int tm   = gw >> 2;
    const int row0 = tm * 16, col0 = tn * 16;

    v8f c = {};
#pragma unroll
    for (int kb = 0; kb < D_MID; kb += 32) {
        v16h a = load_a_frag_h(Hh, D_MID, row0, kb, lane);
        v16h b = load_b_frag_h(W2Th, D_MID, kb, col0, lane);
        c = __builtin_amdgcn_wmma_f32_16x16x32_f16(false, a, false, b,
                                                   (short)0, c, false, false);
    }
    const int mbase = (lane & 16) ? 8 : 0;
    const int n     = lane & 15;
#pragma unroll
    for (int r = 0; r < 8; ++r)
        Z[(size_t)(row0 + mbase + r) * D_EMB + col0 + n] = c[r];
}

// ---------------------------------------------------------------------
// COO SpMM scatter (atomics; edge counts are tiny vs. the rest)
// ---------------------------------------------------------------------
__global__ void zero_f32(float* __restrict__ p, int n) {
    int i = blockIdx.x * blockDim.x + threadIdx.x;
    if (i < n) p[i] = 0.0f;
}

__global__ void spmm_wide(const float* __restrict__ In, float* __restrict__ Out,
                          const float* __restrict__ ev,
                          const int* __restrict__ src,
                          const int* __restrict__ dst) {
    const int e = blockIdx.x;
    const int c = threadIdx.x;
    const float v = ev[e];
    const int s = src[e], d = dst[e];
    atomicAdd(&Out[(size_t)d * D_MID + c], v * In[(size_t)s * D_MID + c]);
}

__global__ void spmm_narrow(const float* __restrict__ In, float* __restrict__ Out,
                            const float* __restrict__ ev,
                            const int* __restrict__ src,
                            const int* __restrict__ dst) {
    const int e = blockIdx.x * 4 + (threadIdx.x >> 6);
    const int c = threadIdx.x & 63;
    const float v = ev[e];
    const int s = src[e], d = dst[e];
    atomicAdd(&Out[(size_t)d * D_EMB + c], v * In[(size_t)s * D_EMB + c]);
}

// ---------------------------------------------------------------------
// Row squared-norms of emb
// ---------------------------------------------------------------------
__global__ void row_sq(const float* __restrict__ emb, float* __restrict__ sq) {
    int i = blockIdx.x * blockDim.x + threadIdx.x;
    if (i < NN) {
        const float* p = emb + (size_t)i * D_EMB;
        float s = 0.0f;
#pragma unroll
        for (int k = 0; k < D_EMB; ++k) s += p[k] * p[k];
        sq[i] = s;
    }
}

// ---------------------------------------------------------------------
// Fused pairwise-distance softmax. Key fact: dist(i,j)=||e_i-e_j||^2 >= 0
// and dist(i,i)=0, so max_j(-dist)=0 exactly -> softmax with a FIXED zero
// shift is exact: out = exp(-dist)/sum_j exp(-dist) + 1e-10. No online max
// tracking needed; pass 1 only accumulates the row sums.
// One workgroup (8 waves) owns a 16-row stripe; emb f16 (1MB) is L2-hot.
// ---------------------------------------------------------------------
__global__ void dist_softmax(const _Float16* __restrict__ embh,
                             const float* __restrict__ sq,
                             float* __restrict__ out) {
    const int lane  = threadIdx.x & 31;
    const int wave  = threadIdx.x >> 5;
    const int row0  = blockIdx.x * 16;
    const int mbase = (lane & 16) ? 8 : 0;
    const int n     = lane & 15;

    __shared__ float red_s[8][32][8];
    __shared__ float fin_s[16];

    // A fragments: constant over both column sweeps
    const v16h a0 = load_a_frag_h(embh, D_EMB, row0, 0,  lane);
    const v16h a1 = load_a_frag_h(embh, D_EMB, row0, 32, lane);

    float sq_m[8];
#pragma unroll
    for (int r = 0; r < 8; ++r) sq_m[r] = sq[row0 + mbase + r];

    float vsum[8];
#pragma unroll
    for (int r = 0; r < 8; ++r) vsum[r] = 0.0f;

    // ---------------- pass 1: row exp-sums ----------------
    for (int j = wave; j < NN / 16; j += 8) {
        const int col0 = j * 16;
        v16h b0 = load_b_frag_h(embh, D_EMB, 0,  col0, lane);
        v16h b1 = load_b_frag_h(embh, D_EMB, 32, col0, lane);
        v8f c = {};
        c = __builtin_amdgcn_wmma_f32_16x16x32_f16(false, a0, false, b0,
                                                   (short)0, c, false, false);
        c = __builtin_amdgcn_wmma_f32_16x16x32_f16(false, a1, false, b1,
                                                   (short)0, c, false, false);
        const float sq_n = sq[col0 + n];
#pragma unroll
        for (int r = 0; r < 8; ++r) {
            float x = 2.0f * c[r] - sq_m[r] - sq_n;   // = -dist <= ~0
            vsum[r] += __expf(x);
        }
    }
#pragma unroll
    for (int r = 0; r < 8; ++r) red_s[wave][lane][r] = vsum[r];
    __syncthreads();

    // plain sum over 8 waves x 16 lanes per row
    if (threadIdx.x < 16) {
        const int M     = threadIdx.x;
        const int lbase = (M < 8) ? 0 : 16;
        const int r     = (M < 8) ? M : M - 8;
        float s = 0.0f;
        for (int w = 0; w < 8; ++w)
            for (int l = 0; l < 16; ++l)
                s += red_s[w][lbase + l][r];
        fin_s[M] = s;
    }
    __syncthreads();

    float rinv[8];
#pragma unroll
    for (int r = 0; r < 8; ++r) rinv[r] = 1.0f / fin_s[mbase + r];

    // ---------------- pass 2: recompute and emit ----------------
    for (int j = wave; j < NN / 16; j += 8) {
        const int col0 = j * 16;
        v16h b0 = load_b_frag_h(embh, D_EMB, 0,  col0, lane);
        v16h b1 = load_b_frag_h(embh, D_EMB, 32, col0, lane);
        v8f c = {};
        c = __builtin_amdgcn_wmma_f32_16x16x32_f16(false, a0, false, b0,
                                                   (short)0, c, false, false);
        c = __builtin_amdgcn_wmma_f32_16x16x32_f16(false, a1, false, b1,
                                                   (short)0, c, false, false);
        const float sq_n = sq[col0 + n];
#pragma unroll
        for (int r = 0; r < 8; ++r) {
            float x = 2.0f * c[r] - sq_m[r] - sq_n;
            out[(size_t)(row0 + mbase + r) * NN + col0 + n] =
                __expf(x) * rinv[r] + 1e-10f;
        }
    }
}

// ---------------------------------------------------------------------
extern "C" void kernel_launch(void* const* d_in, const int* in_sizes, int n_in,
                              void* d_out, int out_size, void* d_ws, size_t ws_size,
                              hipStream_t stream) {
    (void)in_sizes; (void)n_in; (void)out_size; (void)ws_size;

    const float* X   = (const float*)d_in[0];
    const float* W1  = (const float*)d_in[1];
    const float* W2  = (const float*)d_in[2];
    const float* ev  = (const float*)d_in[3];
    const int*   esr = (const int*)d_in[4];
    const int*   eds = (const int*)d_in[5];
    float* out = (float*)d_out;

    // ---- workspace layout ----
    char* ws = (char*)d_ws;
    size_t off = 0;
    float* Y    = (float*)(ws + off); off += (size_t)NN * D_MID * 4;     //  8 MB
    float* H1   = (float*)(ws + off); off += (size_t)NN * D_MID * 4;     //  8 MB
    float* Z    = (float*)(ws + off); off += (size_t)NN * D_EMB * 4;     //  2 MB
    float* EMB  = (float*)(ws + off); off += (size_t)NN * D_EMB * 4;     //  2 MB
    float* SQ   = (float*)(ws + off); off += 64 * 1024;                  // 64 KB (padded)
    _Float16* W1Th = (_Float16*)(ws + off); off += (size_t)D_IN * D_MID * 2;   // 0.5 MB
    _Float16* W2Th = (_Float16*)(ws + off); off += (size_t)D_MID * D_EMB * 2;  // 32 KB
    _Float16* EMBh = (_Float16*)(ws + off); off += (size_t)NN * D_EMB * 2;     //  1 MB
    _Float16* Xh   = (_Float16*)(ws + off);                              // 16 MB (tail)
    _Float16* Hh   = (_Float16*)(ws + off);                              // aliases Xh (disjoint in time)

    // 0) one-time f16 operand preparation (weights stored K-transposed)
    cvt_f32_to_f16<<<(NN * D_IN + 255) / 256, 256, 0, stream>>>(X, Xh, NN * D_IN, 0);
    cvt_transpose_f16<<<D_IN, D_MID, 0, stream>>>(W1, W1Th, D_IN, D_MID);
    cvt_transpose_f16<<<D_MID, D_EMB, 0, stream>>>(W2, W2Th, D_MID, D_EMB);

    // 1) Y = X @ W1                         (WMMA f16->f32)
    gemm_x_w1<<<(512 * 16) / 8, 256, 0, stream>>>(Xh, W1Th, Y);

    // 2) H1 = Laplacian-SpMM(Y)
    zero_f32<<<(NN * D_MID + 255) / 256, 256, 0, stream>>>(H1, NN * D_MID);
    spmm_wide<<<N_EDGES, D_MID, 0, stream>>>(Y, H1, ev, esr, eds);

    // 3) Hh = f16(relu(H1));  Z = Hh @ W2   (WMMA)
    cvt_f32_to_f16<<<(NN * D_MID + 255) / 256, 256, 0, stream>>>(H1, Hh, NN * D_MID, 1);
    gemm_h_w2<<<(512 * 4) / 8, 256, 0, stream>>>(Hh, W2Th, Z);

    // 4) EMB = Laplacian-SpMM(Z)
    zero_f32<<<(NN * D_EMB + 255) / 256, 256, 0, stream>>>(EMB, NN * D_EMB);
    spmm_narrow<<<N_EDGES / 4, 256, 0, stream>>>(Z, EMB, ev, esr, eds);

    // 5) row squared norms (f32) + f16 mirror of emb for WMMA
    row_sq<<<NN / 256, 256, 0, stream>>>(EMB, SQ);
    cvt_f32_to_f16<<<(NN * D_EMB + 255) / 256, 256, 0, stream>>>(EMB, EMBh, NN * D_EMB, 0);

    // 6) fused pairwise-distance + exact zero-shift softmax (WMMA logits)
    dist_softmax<<<NN / 16, 256, 0, stream>>>(EMBh, SQ, out);
}
